// XMR_38242388803806
// MI455X (gfx1250) — matrix operand order repaired and checked by reading, not compiled, not run on hardware
//
#include <hip/hip_runtime.h>
#include <hip/hip_bf16.h>
#include <cstdint>

#define EPSV 1e-5f

typedef __attribute__((ext_vector_type(16))) __bf16 v16bf;
typedef __attribute__((ext_vector_type(8)))  float  v8f;
typedef __attribute__((ext_vector_type(4)))  int    v4i;

union ABfrag { uint4 u[2]; v16bf v; };

#if defined(__AMDGCN__) && __has_builtin(__builtin_amdgcn_global_load_async_to_lds_b128)
#define USE_ASYNC_LDS 1
#else
#define USE_ASYNC_LDS 0
#endif

#if USE_ASYNC_LDS
typedef __attribute__((address_space(1))) v4i* gv4i_ptr;
typedef __attribute__((address_space(3))) v4i* lv4i_ptr;
// 16-byte global -> LDS direct copy (ASYNCcnt-tracked, no VGPR round trip)
__device__ __forceinline__ void async_copy16(const __bf16* g, __bf16* l) {
  __builtin_amdgcn_global_load_async_to_lds_b128((gv4i_ptr)g, (lv4i_ptr)l, 0, 0);
}
__device__ __forceinline__ void wait_async0() {
  asm volatile("s_wait_asynccnt 0" ::: "memory");
}
#endif

// ---------------------------------------------------------------------------
// Pack kernels: f32 -> bf16 (A stays [B][K]; W is transposed to [K][N] so the
// B-tile fill (each thread: one K-row, 16 contiguous N) reads contiguously).
// ---------------------------------------------------------------------------
__global__ void pack_bf16(const float* __restrict__ in, __bf16* __restrict__ out, int n) {
  for (int i = blockIdx.x * blockDim.x + threadIdx.x; i < n; i += gridDim.x * blockDim.x)
    out[i] = (__bf16)in[i];
}

__global__ void pack_w_t(const float* __restrict__ W, __bf16* __restrict__ Wt, int N, int K) {
  int total = N * K;
  for (int i = blockIdx.x * blockDim.x + threadIdx.x; i < total; i += gridDim.x * blockDim.x) {
    int n = i / K, k = i - n * K;
    Wt[(size_t)k * N + n] = (__bf16)W[i];
  }
}

// ---------------------------------------------------------------------------
// Main GEMM: C[2048x2048] = A[2048x3008] * Wt[3008x2048], bf16 in / f32 out.
// 256 threads = 8 waves; block tile 128x128; wave tile 64x32 (4x2 WMMA frags).
// Double-buffered LDS; async global->LDS prefetch of tile k+1 during tile k.
// ---------------------------------------------------------------------------
#define GM 2048
#define GN 2048
#define GK 3008
#define BM 128
#define BN 128
#define BK 32

__global__ __launch_bounds__(256) void gemm_wmma_bf16(
    const __bf16* __restrict__ A, const __bf16* __restrict__ Bt, float* __restrict__ C) {
  __shared__ __align__(16) __bf16 sA[2][BM][BK + 8];  // 2*128*40*2B = 20.0 KB
  __shared__ __align__(16) __bf16 sB[2][BK][BN + 8];  // 2*32*136*2B = 17.0 KB

  const int tid  = threadIdx.x;
  const int lane = tid & 31;
  const int wave = tid >> 5;
  const int h    = lane >> 4;   // half-wave select
  const int ml   = lane & 15;
  const int bm   = blockIdx.y * BM;
  const int bn   = blockIdx.x * BN;
  const int wm   = (wave >> 2) * 64;  // 2 waves along M
  const int wn   = (wave & 3) * 32;   // 4 waves along N

  v8f acc[4][2] = {};

  const int arow = tid >> 1, acol = (tid & 1) * 16;  // A tile: 128x32 bf16
  const int brow = tid >> 3, bcol = (tid & 7) * 16;  // B tile: 32x128 bf16

  const __bf16* gaBase = A + (size_t)(bm + arow) * GK + acol;
  const __bf16* gbBase = Bt + (size_t)brow * GN + bn + bcol;

  // tile fill: global (k0) -> LDS buffer p
  auto loadTiles = [&](int k0, int p) {
#if USE_ASYNC_LDS
    const __bf16* ga = gaBase + k0;
    async_copy16(ga,     &sA[p][arow][acol]);
    async_copy16(ga + 8, &sA[p][arow][acol + 8]);
    const __bf16* gb = gbBase + (size_t)k0 * GN;
    async_copy16(gb,     &sB[p][brow][bcol]);
    async_copy16(gb + 8, &sB[p][brow][bcol + 8]);
#else
    const __bf16* ga = gaBase + k0;
    *(uint4*)&sA[p][arow][acol]     = *(const uint4*)(ga);
    *(uint4*)&sA[p][arow][acol + 8] = *(const uint4*)(ga + 8);
    const __bf16* gb = gbBase + (size_t)k0 * GN;
    *(uint4*)&sB[p][brow][bcol]     = *(const uint4*)(gb);
    *(uint4*)&sB[p][brow][bcol + 8] = *(const uint4*)(gb + 8);
#endif
  };

  // prologue: tile 0 into buffer 0
  loadTiles(0, 0);
#if USE_ASYNC_LDS
  wait_async0();
#endif
  __syncthreads();

  const int itEnd = GK / BK;  // 94
  for (int it = 0; it < itEnd; ++it) {
    const int p = it & 1;
    // prefetch next tile into the buffer that was fully consumed last iter
    if (it + 1 < itEnd) loadTiles((it + 1) * BK, p ^ 1);

    // A frag (16x32, ISA 7.12.2): lane m=ml, halves i<8 -> K=i+8h, i>=8 -> K=i+8+8h
    v16bf afr[4];
#pragma unroll
    for (int mi = 0; mi < 4; mi++) {
      ABfrag t;
      t.u[0] = *(const uint4*)&sA[p][wm + mi * 16 + ml][8 * h];
      t.u[1] = *(const uint4*)&sA[p][wm + mi * 16 + ml][8 * h + 16];
      afr[mi] = t.v;
    }
    // B frag (32x16): lane n=ml, halves i<8 -> K=i+8h, i>=8 -> K=i+8+8h
    v16bf bfr[2];
#pragma unroll
    for (int ni = 0; ni < 2; ni++) {
      ABfrag t;
#pragma unroll
      for (int s = 0; s < 16; s++)
        t.v[s] = sB[p][(s < 8 ? 8 * h + s : 16 + 8 * h + (s - 8))][wn + ni * 16 + ml];
      bfr[ni] = t.v;
    }
#pragma unroll
    for (int mi = 0; mi < 4; mi++)
#pragma unroll
      for (int ni = 0; ni < 2; ni++)
        acc[mi][ni] = __builtin_amdgcn_wmma_f32_16x16x32_bf16(
            false, afr[mi], false, bfr[ni], (short)0, acc[mi][ni], false, false);

#if USE_ASYNC_LDS
    wait_async0();          // next tile fully landed in LDS (this wave's share)
#endif
    __syncthreads();        // all waves' shares landed; frag reads of p are done
  }

  // C/D layout: VGPR r -> row r + 8*h, col = ml
#pragma unroll
  for (int mi = 0; mi < 4; mi++)
#pragma unroll
    for (int ni = 0; ni < 2; ni++)
#pragma unroll
      for (int r = 0; r < 8; r++) {
        int row = bm + wm + mi * 16 + r + 8 * h;
        int col = bn + wn + ni * 16 + ml;
        C[(size_t)row * GN + col] = acc[mi][ni][r];
      }
}

// ---------------------------------------------------------------------------
// GO hierarchy layer: y=tanh(x@W^T) per term t, plus batch mean/var (one block
// owns one term -> no atomics). T buffer stores tanh activations.
// ---------------------------------------------------------------------------
__global__ __launch_bounds__(256) void go_layer_fwd(
    const float* __restrict__ X, const float* __restrict__ W,
    float* __restrict__ T, float* __restrict__ stats, int Tl, int cin) {
  const int t = blockIdx.x;
  const int tid = threadIdx.x;
  __shared__ float wsm[8 * 32];
  __shared__ float red[256 * 16];
  for (int i = tid; i < 8 * cin; i += 256) wsm[i] = W[(size_t)t * 8 * cin + i];
  __syncthreads();
  float s[8], s2[8];
#pragma unroll
  for (int j = 0; j < 8; j++) { s[j] = 0.f; s2[j] = 0.f; }
  for (int b = tid; b < 2048; b += 256) {
    const float* xr = X + (size_t)b * Tl * cin + (size_t)t * cin;
#pragma unroll
    for (int hh = 0; hh < 8; hh++) {
      float acc = 0.f;
      for (int i = 0; i < cin; i++) acc += xr[i] * wsm[hh * cin + i];
      float th = tanhf(acc);
      T[((size_t)b * Tl + t) * 8 + hh] = th;
      s[hh] += th;
      s2[hh] += th * th;
    }
  }
#pragma unroll
  for (int j = 0; j < 8; j++) { red[tid * 16 + j] = s[j]; red[tid * 16 + 8 + j] = s2[j]; }
  __syncthreads();
  for (int st = 128; st > 0; st >>= 1) {
    if (tid < st)
      for (int j = 0; j < 16; j++) red[tid * 16 + j] += red[(tid + st) * 16 + j];
    __syncthreads();
  }
  if (tid < 8) {
    float mu  = red[tid] * (1.f / 2048.f);
    float ex2 = red[8 + tid] * (1.f / 2048.f);
    stats[t * 8 + tid] = mu;
    stats[Tl * 8 + t * 8 + tid] = ex2 - mu * mu;
  }
}

__global__ void go_norm(const float* __restrict__ T, const float* __restrict__ stats,
                        const float* __restrict__ g, const float* __restrict__ bb,
                        float* __restrict__ out, float* __restrict__ out2,
                        int Tl, int n) {
  int idx = blockIdx.x * blockDim.x + threadIdx.x;
  if (idx >= n) return;
  int hh = idx & 7;
  int t  = (idx >> 3) % Tl;
  float mu = stats[t * 8 + hh], var = stats[Tl * 8 + t * 8 + hh];
  float o = g[t * 8 + hh] * (T[idx] - mu) * rsqrtf(var + EPSV) + bb[t * 8 + hh];
  out[idx] = o;
  if (out2) out2[idx] = o;
}

// ---------------------------------------------------------------------------
// Whole drug branch fused, one block per molecule (all intermediates in LDS):
// hv = relu(embed[fp] @ Wfp^T + bfp); hs = hv + adj@hv; mol = sum_i hs_i/|hs_i|;
// v = relu(mol@Wout^T + bout); drug = v@Wprop^T + bprop.
// ---------------------------------------------------------------------------
__global__ __launch_bounds__(256) void drug_kernel(
    const int* __restrict__ fp, const float* __restrict__ adj,
    const float* __restrict__ embed,
    const float* __restrict__ Wfp, const float* __restrict__ bfp,
    const float* __restrict__ Wout, const float* __restrict__ bout,
    const float* __restrict__ Wprop, const float* __restrict__ bprop,
    float* __restrict__ drug) {
  const int b = blockIdx.x;
  const int tid = threadIdx.x;
  __shared__ float s_hv[32][64];
  __shared__ float s_hs[32][65];  // pad: norm pass reads a row per lane
  __shared__ float s_adj[32][32];
  __shared__ float s_nrm[32];
  __shared__ float s_mol[64];
  __shared__ float s_v[64];

  for (int i = tid; i < 32 * 32; i += 256) s_adj[i >> 5][i & 31] = adj[(size_t)b * 1024 + i];
  for (int e = tid; e < 2048; e += 256) {
    int a = e >> 6, d = e & 63;
    const float* fv = embed + (size_t)fp[b * 32 + a] * 64;
    const float* wr = Wfp + d * 64;
    float acc = bfp[d];
    for (int j = 0; j < 64; j++) acc += fv[j] * wr[j];
    s_hv[a][d] = fmaxf(acc, 0.f);
  }
  __syncthreads();
  for (int e = tid; e < 2048; e += 256) {
    int i = e >> 6, d = e & 63;
    float acc = s_hv[i][d];
    for (int j = 0; j < 32; j++) acc += s_adj[i][j] * s_hv[j][d];
    s_hs[i][d] = acc;
  }
  __syncthreads();
  if (tid < 32) {
    float acc = 0.f;
    for (int d = 0; d < 64; d++) { float x = s_hs[tid][d]; acc += x * x; }
    s_nrm[tid] = fmaxf(sqrtf(acc), 1e-12f);
  }
  __syncthreads();
  if (tid < 64) {
    float acc = 0.f;
    for (int i = 0; i < 32; i++) acc += s_hs[i][tid] / s_nrm[i];
    s_mol[tid] = acc;
  }
  __syncthreads();
  if (tid < 64) {
    const float* wr = Wout + tid * 64;
    float acc = bout[tid];
    for (int j = 0; j < 64; j++) acc += s_mol[j] * wr[j];
    s_v[tid] = fmaxf(acc, 0.f);
  }
  __syncthreads();
  if (tid < 64) {
    const float* wr = Wprop + tid * 64;
    float acc = bprop[tid];
    for (int j = 0; j < 64; j++) acc += s_v[j] * wr[j];
    drug[(size_t)b * 64 + tid] = acc;
  }
}

// ---------------------------------------------------------------------------
// Final head: z = tanh([root,drug] @ Wfin^T + bfin), batch stats per output.
// ---------------------------------------------------------------------------
__global__ __launch_bounds__(256) void fin_z(
    const float* __restrict__ root, const float* __restrict__ drug,
    const float* __restrict__ Wfin, const float* __restrict__ bfin,
    float* __restrict__ z, float* __restrict__ statsF) {
  const int ho = blockIdx.x;  // 0..63
  const int tid = threadIdx.x;
  __shared__ float w[72];
  __shared__ float rs[256], rs2[256];
  for (int i = tid; i < 72; i += 256) w[i] = Wfin[ho * 72 + i];
  __syncthreads();
  float s = 0.f, s2 = 0.f;
  for (int b = tid; b < 2048; b += 256) {
    float acc = bfin[ho];
    const float* r = root + (size_t)b * 8;
    for (int i = 0; i < 8; i++) acc += r[i] * w[i];
    const float* dd = drug + (size_t)b * 64;
    for (int i = 0; i < 64; i++) acc += dd[i] * w[8 + i];
    float zz = tanhf(acc);
    z[(size_t)b * 64 + ho] = zz;
    s += zz;
    s2 += zz * zz;
  }
  rs[tid] = s; rs2[tid] = s2;
  __syncthreads();
  for (int st = 128; st > 0; st >>= 1) {
    if (tid < st) { rs[tid] += rs[tid + st]; rs2[tid] += rs2[tid + st]; }
    __syncthreads();
  }
  if (tid == 0) {
    float mu = rs[0] * (1.f / 2048.f);
    statsF[ho] = mu;
    statsF[64 + ho] = rs2[0] * (1.f / 2048.f) - mu * mu;
  }
}

__global__ void fin_out(const float* __restrict__ z, const float* __restrict__ statsF,
                        const float* __restrict__ g, const float* __restrict__ bb,
                        const float* __restrict__ Waux, const float* __restrict__ baux,
                        const float* __restrict__ Wo, const float* __restrict__ bo,
                        float* __restrict__ aux_final, float* __restrict__ outf) {
  int b = blockIdx.x * blockDim.x + threadIdx.x;
  if (b >= 2048) return;
  float dot = 0.f;
  for (int ho = 0; ho < 64; ho++) {
    float mu = statsF[ho], var = statsF[64 + ho];
    float o = g[ho] * (z[(size_t)b * 64 + ho] - mu) * rsqrtf(var + EPSV) + bb[ho];
    outf[(size_t)b * 64 + ho] = o;
    dot += o * Waux[ho];
  }
  float auxf = tanhf(dot + baux[0]);
  aux_final[b] = auxf * Wo[0] + bo[0];
}

// ---------------------------------------------------------------------------
extern "C" void kernel_launch(void* const* d_in, const int* in_sizes, int n_in,
                              void* d_out, int out_size, void* d_ws, size_t ws_size,
                              hipStream_t stream) {
  (void)in_sizes; (void)n_in; (void)out_size; (void)ws_size;
  // setup_inputs() insertion order flattening:
  const float* gene  = (const float*)d_in[0];
  const int*   fp    = (const int*)d_in[1];
  const float* adj   = (const float*)d_in[2];
  const float* Wgene = (const float*)d_in[3];
  const float* Wgo[5]; const float* bng[5]; const float* bnb[5];
  for (int l = 0; l < 5; l++) {
    Wgo[l] = (const float*)d_in[4 + l];
    bng[l] = (const float*)d_in[9 + l];
    bnb[l] = (const float*)d_in[14 + l];
  }
  // d_in[19..38] = aux1_w/aux1_b/aux2_w/aux2_b (not in returned outputs)
  const float* embed = (const float*)d_in[39];
  const float* Wfp   = (const float*)d_in[40];
  const float* bfp   = (const float*)d_in[41];
  const float* Wout  = (const float*)d_in[42];
  const float* bout  = (const float*)d_in[43];
  const float* Wprop = (const float*)d_in[44];
  const float* bprop = (const float*)d_in[45];
  const float* Wfin  = (const float*)d_in[46];
  const float* bfin  = (const float*)d_in[47];
  const float* bnfg  = (const float*)d_in[48];
  const float* bnfb  = (const float*)d_in[49];
  const float* Waux  = (const float*)d_in[50];
  const float* baux  = (const float*)d_in[51];
  const float* Wo    = (const float*)d_in[52];
  const float* bo    = (const float*)d_in[53];

  char* ws = (char*)d_ws;
  size_t off = 0;
  auto take = [&](size_t bytes) -> char* {
    char* p = ws + off;
    off += (bytes + 255) & ~(size_t)255;
    return p;
  };
  __bf16* Abf    = (__bf16*)take((size_t)2048 * 3008 * 2);
  __bf16* Wt     = (__bf16*)take((size_t)3008 * 2048 * 2);
  float*  X0     = (float*)take((size_t)2048 * 2048 * 4);
  float*  Tb     = (float*)take((size_t)2048 * 1024 * 4);
  float*  O1     = (float*)take((size_t)2048 * 1024 * 4);
  float*  O2     = (float*)take((size_t)2048 * 1024 * 4);
  float*  stats  = (float*)take(2048 * 2 * 4);
  float*  drugb  = (float*)take((size_t)2048 * 64 * 4);
  float*  zb     = (float*)take((size_t)2048 * 64 * 4);
  float*  statsF = (float*)take(512);

  float* aux_final = (float*)d_out;                    // (2048,1)
  float* outf      = (float*)d_out + 2048;             // (2048,64)
  float* root      = (float*)d_out + 2048 + 2048 * 64; // (2048,8)

  // 1) pack to bf16 (+ W transpose to K-major)
  pack_bf16<<<2048, 256, 0, stream>>>(gene, Abf, 2048 * 3008);
  pack_w_t<<<2048, 256, 0, stream>>>(Wgene, Wt, 2048, 3008);

  // 2) dominant GEMM via WMMA bf16 with async double-buffered LDS pipeline
  gemm_wmma_bf16<<<dim3(GN / BN, GM / BM), 256, 0, stream>>>(Abf, Wt, X0);

  // 3) GO hierarchy (reshape between levels is a no-op on flat layout)
  const int TLs[5]  = {128, 32, 8, 2, 1};
  const int cins[5] = {16, 32, 32, 32, 16};
  const float* xin = X0;
  float* outs[5] = {O1, O2, O1, O2, O1};
  for (int l = 0; l < 5; l++) {
    go_layer_fwd<<<TLs[l], 256, 0, stream>>>(xin, Wgo[l], Tb, stats, TLs[l], cins[l]);
    int nelem = 2048 * TLs[l] * 8;
    float* o2 = (l == 4) ? root : nullptr;  // last level's out == root_out
    go_norm<<<(nelem + 255) / 256, 256, 0, stream>>>(Tb, stats, bng[l], bnb[l],
                                                     outs[l], o2, TLs[l], nelem);
    xin = outs[l];
  }

  // 4) fused drug branch (one block per molecule)
  drug_kernel<<<2048, 256, 0, stream>>>(fp, adj, embed, Wfp, bfp, Wout, bout,
                                        Wprop, bprop, drugb);

  // 5) final head + batchnorm + aux output
  fin_z<<<64, 256, 0, stream>>>(root, drugb, Wfin, bfin, zb, statsF);
  fin_out<<<8, 256, 0, stream>>>(zb, statsF, bnfg, bnfb, Waux, baux, Wo, bo,
                                 aux_final, outf);
}
